// Coarse_Grain_3DLayer_10084583211475
// MI455X (gfx1250) — matrix operand order repaired and checked by reading, not compiled, not run on hardware
//
#include <hip/hip_runtime.h>
#include <hip/hip_bf16.h>
#include <math.h>

// Problem constants (match reference)
#define NN 50000
#define NB_ 256
#define DD 128
#define FF 16
#define EE 800000
#define NRBF 15
#define EPS_ 1e-7f
#define NEG_ 0.01f
#define MAXLEN_ 5.0f

typedef __attribute__((ext_vector_type(16))) _Float16 v16h;
typedef __attribute__((ext_vector_type(8)))  _Float16 h8;
typedef __attribute__((ext_vector_type(8)))  float    v8f;

static __device__ __forceinline__ v8f wmma_f16(v16h a, v16h b, v8f c) {
  // D = A(16x32 f16) * B(32x16 f16) + C(16x16 f32)
  return __builtin_amdgcn_wmma_f32_16x16x32_f16(false, a, false, b, (short)0, c,
                                                false, false);
}

// ---- WMMA fragment loaders (CDNA5 16-bit layouts, ISA 7.12.2) ----
// A fragment element map: a[0..7]  = A[row][k0 + half*8 + 0..7]
//                         a[8..15] = A[row][k0 + 16 + half*8 + 0..7]
// B fragment element map: b[0..7]  = B[col][k0 + half*8 + 0..7]
//                         b[8..15] = B[col][k0 + 16 + half*8 + 0..7]  (B stored col-major)

// f32 A operand, concat(A1[:,0:K1], A2[:,0:KTOT-K1]); K1, KTOT multiples of 16
// => per-16-group source selection is compile-time uniform.
template <int KTOT, int K1, int LDA1, int LDA2>
static __device__ __forceinline__ v16h load_a_f32(const float* A1, const float* A2,
                                                  int row, int k0, int half) {
#pragma clang diagnostic push
  v16h a;
#pragma unroll
  for (int g = 0; g < 2; ++g) {
    const int kG = k0 + g * 16;  // compile-time after unrolling
    float4 x0, x1;
    if (kG + 16 <= K1) {
      const float4* p = (const float4*)(A1 + (size_t)row * LDA1 + kG + half * 8);
      x0 = p[0]; x1 = p[1];
    } else if (kG >= K1 && kG + 16 <= KTOT) {
      const float4* p = (const float4*)(A2 + (size_t)row * LDA2 + (kG - K1) + half * 8);
      x0 = p[0]; x1 = p[1];
    } else {
      x0 = make_float4(0.f, 0.f, 0.f, 0.f); x1 = x0;
    }
    a[g * 8 + 0] = (_Float16)x0.x; a[g * 8 + 1] = (_Float16)x0.y;
    a[g * 8 + 2] = (_Float16)x0.z; a[g * 8 + 3] = (_Float16)x0.w;
    a[g * 8 + 4] = (_Float16)x1.x; a[g * 8 + 5] = (_Float16)x1.y;
    a[g * 8 + 6] = (_Float16)x1.z; a[g * 8 + 7] = (_Float16)x1.w;
  }
#pragma clang diagnostic pop
  return a;
}

// f16 A operand (global or LDS), row-major stride LDA
template <int LDA>
static __device__ __forceinline__ v16h load_a_h16(const _Float16* A, int row,
                                                  int k0, int half) {
  const h8* p0 = (const h8*)(A + (size_t)row * LDA + k0 + half * 8);
  const h8* p1 = (const h8*)(A + (size_t)row * LDA + k0 + 16 + half * 8);
  h8 x0 = *p0, x1 = *p1;
  v16h a;
#pragma unroll
  for (int e = 0; e < 8; ++e) { a[e] = x0[e]; a[8 + e] = x1[e]; }
  return a;
}

// f16 B operand from packed col-major weights Wt[col][KPAD]
template <int KPAD>
static __device__ __forceinline__ v16h load_b_h16(const _Float16* Wt, int col,
                                                  int k0, int half) {
  const h8* p0 = (const h8*)(Wt + (size_t)col * KPAD + k0 + half * 8);
  const h8* p1 = (const h8*)(Wt + (size_t)col * KPAD + k0 + 16 + half * 8);
  h8 x0 = *p0, x1 = *p1;
  v16h b;
#pragma unroll
  for (int e = 0; e < 8; ++e) { b[e] = x0[e]; b[8 + e] = x1[e]; }
  return b;
}

// ---------------- weight packing ----------------

// W[K][Dout] row-major  ->  Wt[Dout][KPAD] f16 (transpose + pad)
__global__ void pack_wT_kernel(const float* W, _Float16* Wt, int K, int Dout, int KPAD) {
  int i = blockIdx.x * blockDim.x + threadIdx.x;
  if (i >= Dout * KPAD) return;
  int d = i / KPAD, k = i % KPAD;
  Wt[i] = (k < K) ? (_Float16)W[(size_t)k * Dout + d] : (_Float16)0.f;
}

// W[Dout][K] row-major  ->  Wt[Dout][KPAD] f16 (pad only)
__global__ void pack_w_kernel(const float* W, _Float16* Wt, int K, int Dout, int KPAD) {
  int i = blockIdx.x * blockDim.x + threadIdx.x;
  if (i >= Dout * KPAD) return;
  int d = i / KPAD, k = i % KPAD;
  Wt[i] = (k < K) ? (_Float16)W[(size_t)d * K + k] : (_Float16)0.f;
}

// ---------------- kernels ----------------

__global__ void zero_kernel(float* agg_h, float* agg_v, float* cnt) {
  int i = blockIdx.x * blockDim.x + threadIdx.x;
  if (i < NN * DD) agg_h[i] = 0.f;
  if (i < NN * FF * 3) agg_v[i] = 0.f;
  if (i < NN) cnt[i] = 0.f;
}

// Per-node vector-neuron layers vn1/vn2 (norms) and vn3 (vectors)
__global__ void node_vn_kernel(const float* vA,
                               const float* W1, const float* U1,
                               const float* W2, const float* U2,
                               const float* W3, const float* U3,
                               float* vnorm1, float* vnorm2, float* vr3) {
  __shared__ float sW[6][16][16];
  {
    const float* srcp[6] = {W1, U1, W2, U2, W3, U3};
    for (int i = threadIdx.x; i < 6 * 256; i += blockDim.x)
      sW[i >> 8][(i >> 4) & 15][i & 15] = srcp[i >> 8][i & 255];
  }
  __syncthreads();
  int n = blockIdx.x * blockDim.x + threadIdx.x;
  if (n >= NN) return;
  float v[16][3];
#pragma unroll
  for (int f = 0; f < 16; ++f)
#pragma unroll
    for (int c = 0; c < 3; ++c)
      v[f][c] = vA[(size_t)n * 48 + f * 3 + c];

#pragma unroll
  for (int mi = 0; mi < 3; ++mi) {
    const float (*W)[16] = sW[2 * mi];
    const float (*U)[16] = sW[2 * mi + 1];
    for (int g = 0; g < 16; ++g) {
      float q0 = 0, q1 = 0, q2 = 0, k0 = 0, k1 = 0, k2 = 0;
#pragma unroll
      for (int f = 0; f < 16; ++f) {
        float w = W[g][f], u = U[g][f];
        q0 += w * v[f][0]; q1 += w * v[f][1]; q2 += w * v[f][2];
        k0 += u * v[f][0]; k1 += u * v[f][1]; k2 += u * v[f][2];
      }
      float dot = q0 * k0 + q1 * k1 + q2 * k2;
      float ksq = k0 * k0 + k1 * k1 + k2 * k2 + EPS_;
      if (dot < 0.f) { float s = dot / ksq; q0 -= s * k0; q1 -= s * k1; q2 -= s * k2; }
      if (mi == 2) {
        vr3[(size_t)n * 48 + g * 3 + 0] = q0;
        vr3[(size_t)n * 48 + g * 3 + 1] = q1;
        vr3[(size_t)n * 48 + g * 3 + 2] = q2;
      } else {
        float nr = sqrtf(q0 * q0 + q1 * q1 + q2 * q2) + EPS_;
        (mi == 0 ? vnorm1 : vnorm2)[(size_t)n * 16 + g] = nr;
      }
    }
  }
}

// Fully specialized WMMA GEMM: Out = act(concat(A1,A2) @ W + bias)
// OMODE: 0 = f32 row-major, 1 = f16 row-major, 2 = f16 transposed (LDO = #rows)
template <int KTOT, int K1, int LDA1, int LDA2, int DOUT, int LDO,
          bool HAS_BIAS, bool ACT, int OMODE>
__global__ void gemm_t(const float* A1, const float* A2, const _Float16* Wt,
                       const float* bias, void* OutP) {
  constexpr int KPAD = (KTOT + 31) & ~31;
  int row0 = blockIdx.x * 16, col0 = blockIdx.y * 16;
  int l = threadIdx.x & 31;
  int n = l & 15, half = l >> 4;
  v8f c = {};
#pragma unroll
  for (int k0 = 0; k0 < KTOT; k0 += 32) {
    v16h a = load_a_f32<KTOT, K1, LDA1, LDA2>(A1, A2, row0 + n, k0, half);
    v16h b = load_b_h16<KPAD>(Wt, col0 + n, k0, half);
    c = wmma_f16(a, b, c);
  }
#pragma unroll
  for (int v = 0; v < 8; ++v) {
    int m = v + (half << 3);
    float x = c[v];
    if (HAS_BIAS) x += bias[col0 + n];
    if (ACT) x = x >= 0.f ? x : NEG_ * x;
    if (OMODE == 0)
      ((float*)OutP)[(size_t)(row0 + m) * LDO + col0 + n] = x;
    else if (OMODE == 1)
      ((_Float16*)OutP)[(size_t)(row0 + m) * LDO + col0 + n] = (_Float16)x;
    else
      ((_Float16*)OutP)[(size_t)(col0 + n) * LDO + row0 + m] = (_Float16)x;
  }
}

__global__ void vp_mul_kernel(const float* vr3, const float* s3, float* vp) {
  int i = blockIdx.x * blockDim.x + threadIdx.x;
  if (i >= NN * 48) return;
  int n = i / 48; int f = (i % 48) / 3;
  vp[i] = vr3[i] * s3[(size_t)n * 16 + f];
}

// Cross-attention: 16 query rows per block; WMMA scores -> softmax -> WMMA out
__global__ void attn_kernel(const _Float16* q16, const _Float16* kk16,
                            const _Float16* vvT16, const float* mask, float* attn) {
  __shared__ float s[16][256];
  __shared__ _Float16 P[16][256];
  int row0 = blockIdx.x * 16;
  int wv = threadIdx.x >> 5;
  int l = threadIdx.x & 31;
  int n = l & 15, half = l >> 4;

  // scores = q @ kk^T : each wave covers 2 tiles of 16 cols
#pragma unroll
  for (int t = 0; t < 2; ++t) {
    int col0 = wv * 32 + t * 16;
    v8f c = {};
#pragma unroll
    for (int k0 = 0; k0 < DD; k0 += 32) {
      v16h a = load_a_h16<DD>(q16, row0 + n, k0, half);
      v16h b = load_b_h16<DD>(kk16, col0 + n, k0, half);
      c = wmma_f16(a, b, c);
    }
#pragma unroll
    for (int v = 0; v < 8; ++v) s[v + (half << 3)][col0 + n] = c[v];
  }
  __syncthreads();

  if (threadIdx.x < 16) {
    int m = threadIdx.x;
    const float* mrow = mask + (size_t)(row0 + m) * NB_;
    float mx = -3.4e38f;
    for (int j = 0; j < NB_; ++j) {
      float mk = mrow[j];
      float a = mk * s[m][j] - 1000.f * (1.f - mk);
      s[m][j] = a;
      mx = fmaxf(mx, a);
    }
    float sum = 0.f;
    for (int j = 0; j < NB_; ++j) { float e = expf(s[m][j] - mx); s[m][j] = e; sum += e; }
    float inv = 1.f / sum;
    for (int j = 0; j < NB_; ++j) P[m][j] = (_Float16)(s[m][j] * inv);
  }
  __syncthreads();

  {
    int col0 = wv * 16;
    v8f c = {};
#pragma unroll
    for (int k0 = 0; k0 < NB_; k0 += 32) {
      v16h a = load_a_h16<256>(&P[0][0], n, k0, half);
      v16h b = load_b_h16<NB_>(vvT16, col0 + n, k0, half);
      c = wmma_f16(a, b, c);
    }
#pragma unroll
    for (int v = 0; v < 8; ++v)
      attn[(size_t)(row0 + v + (half << 3)) * DD + col0 + n] = c[v];
  }
}

// Fused edge pass: RBF -> messages -> atomic scatter (1 wave = 1 edge)
__global__ void edge_kernel(const float* xA, const int* src, const int* dst,
                            const float* rbfk,
                            const float* W1, const float* b1,
                            const float* W2, const float* b2,
                            const float* W3, const float* b3,
                            const float* hp, const float* hpp, const float* vp,
                            float* agg_h, float* agg_v, float* cnt) {
  __shared__ float sW1[NRBF][DD];
  __shared__ float sW2[NRBF][FF];
  __shared__ float sW3[NRBF][FF];
  __shared__ float sb1[DD], sb2[FF], sb3[FF];
  for (int i = threadIdx.x; i < NRBF * DD; i += 256) sW1[i / DD][i % DD] = W1[i];
  for (int i = threadIdx.x; i < NRBF * FF; i += 256) {
    sW2[i / FF][i % FF] = W2[i];
    sW3[i / FF][i % FF] = W3[i];
  }
  if (threadIdx.x < DD) sb1[threadIdx.x] = b1[threadIdx.x];
  if (threadIdx.x < FF) { sb2[threadIdx.x] = b2[threadIdx.x]; sb3[threadIdx.x] = b3[threadIdx.x]; }
  __syncthreads();

  int e = blockIdx.x * 8 + (threadIdx.x >> 5);
  if (e >= EE) return;
  int lane = threadIdx.x & 31;
  int si = src[e], di = dst[e];
  float r0 = xA[si * 3 + 0] - xA[di * 3 + 0];
  float r1 = xA[si * 3 + 1] - xA[di * 3 + 1];
  float r2 = xA[si * 3 + 2] - xA[di * 3 + 2];
  float rn = sqrtf(r0 * r0 + r1 * r1 + r2 * r2);
  float kc = rbfk[0];
  float rbf[NRBF];
#pragma unroll
  for (int rI = 0; rI < NRBF; ++rI) {
    float d = rn - (MAXLEN_ / (NRBF - 1)) * rI;
    rbf[rI] = expf(-d * d * kc);
  }

  // scalar message: 128 channels, 4 per lane (coalesced c = i*32 + lane)
  size_t hb = (size_t)di * DD;
#pragma unroll
  for (int i = 0; i < 4; ++i) {
    int c = i * 32 + lane;
    float acc = sb1[c];
#pragma unroll
    for (int rI = 0; rI < NRBF; ++rI) acc += rbf[rI] * sW1[rI][c];
    atomicAdd(&agg_h[hb + c], acc * hp[hb + c]);
  }
  // vector message: lanes 0..15 take f = lane
  if (lane < FF) {
    int f = lane;
    float a1 = sb2[f], a2 = sb3[f];
#pragma unroll
    for (int rI = 0; rI < NRBF; ++rI) { a1 += rbf[rI] * sW2[rI][f]; a2 += rbf[rI] * sW3[rI][f]; }
    a2 *= hpp[(size_t)di * FF + f];
    size_t vb = (size_t)di * 48 + f * 3;
    atomicAdd(&agg_v[vb + 0], a1 * vp[vb + 0] + a2 * r0);
    atomicAdd(&agg_v[vb + 1], a1 * vp[vb + 1] + a2 * r1);
    atomicAdd(&agg_v[vb + 2], a1 * vp[vb + 2] + a2 * r2);
  }
  if (lane == 0) atomicAdd(&cnt[di], 1.0f);
}

__global__ void norm_agg_kernel(float* agg_h, float* agg_v, const float* cnt) {
  int i = blockIdx.x * blockDim.x + threadIdx.x;
  const int nh = NN * DD;
  const int tot = nh + NN * 48;
  if (i >= tot) return;
  if (i < nh) {
    float den = fmaxf(cnt[i >> 7], 1.0f);
    agg_h[i] /= den;
  } else {
    int j = i - nh;
    float den = fmaxf(cnt[j / 48], 1.0f);
    agg_v[j] /= den;
  }
}

// Fully fused GRU: gi and gh WMMA tiles in LDS, gate math in-block
__global__ void gru_kernel(const float* hA, const float* attn, const float* agg_h,
                           const _Float16* Wih16, const _Float16* Whh16,
                           const float* bih, const float* bhh, float* out_h) {
  __shared__ float gi[16][384];
  __shared__ float gh[16][384];
  int row0 = blockIdx.x * 16;
  int wv = threadIdx.x >> 5;
  int l = threadIdx.x & 31;
  int n = l & 15, half = l >> 4;

#pragma unroll
  for (int cb = 0; cb < 3; ++cb) {
    int col0 = cb * 128 + wv * 16;
    v8f ci = {};
#pragma unroll
    for (int k0 = 0; k0 < 256; k0 += 32) {          // x_in = [h_A, attn]
      v16h a = load_a_f32<256, 128, 128, 128>(hA, attn, row0 + n, k0, half);
      v16h b = load_b_h16<256>(Wih16, col0 + n, k0, half);   // gi = x_in @ Wih^T
      ci = wmma_f16(a, b, ci);
    }
    v8f ch = {};
#pragma unroll
    for (int k0 = 0; k0 < DD; k0 += 32) {
      v16h a = load_a_f32<128, 128, 128, 128>(agg_h, agg_h, row0 + n, k0, half);
      v16h b = load_b_h16<128>(Whh16, col0 + n, k0, half);   // gh = agg_h @ Whh^T
      ch = wmma_f16(a, b, ch);
    }
#pragma unroll
    for (int v = 0; v < 8; ++v) {
      int m = v + (half << 3);
      gi[m][col0 + n] = ci[v] + bih[col0 + n];
      gh[m][col0 + n] = ch[v] + bhh[col0 + n];
    }
  }
  __syncthreads();

  for (int idx = threadIdx.x; idx < 16 * DD; idx += 256) {
    int m = idx >> 7, d = idx & 127;
    float r = 1.f / (1.f + expf(-(gi[m][d] + gh[m][d])));
    float z = 1.f / (1.f + expf(-(gi[m][128 + d] + gh[m][128 + d])));
    float ng = tanhf(gi[m][256 + d] + r * gh[m][256 + d]);
    size_t o = (size_t)(row0 + m) * DD + d;
    float h = hA[o];
    out_h[o] = h + ((1.f - z) * ng + z * agg_h[o]);
  }
}

// Vector MLP with residual: one wave per node, vh staged in LDS
__global__ void vmlp_kernel(const float* vA, const float* agg_v,
                            const float* W1, const float* U1, const float* W2,
                            float* out_v) {
  __shared__ float sW1[32][32], sU1[32][32], sW2[16][32];
  __shared__ float sVin[8][32][3], sVh[8][32][3];
  for (int i = threadIdx.x; i < 1024; i += blockDim.x) {
    sW1[i >> 5][i & 31] = W1[i];
    sU1[i >> 5][i & 31] = U1[i];
  }
  for (int i = threadIdx.x; i < 512; i += blockDim.x) sW2[i >> 5][i & 31] = W2[i];
  __syncthreads();

  int wv = threadIdx.x >> 5, lane = threadIdx.x & 31;
  int nId = blockIdx.x * 8 + wv;
  bool ok = nId < NN;
  if (ok) {
#pragma unroll
    for (int c = 0; c < 3; ++c)
      sVin[wv][lane][c] = (lane < 16)
          ? vA[(size_t)nId * 48 + lane * 3 + c]
          : agg_v[(size_t)nId * 48 + (lane - 16) * 3 + c];
  }
  __syncthreads();
  if (ok) {
    int g = lane;
    float q0 = 0, q1 = 0, q2 = 0, k0 = 0, k1 = 0, k2 = 0;
#pragma unroll
    for (int f = 0; f < 32; ++f) {
      float w = sW1[g][f], u = sU1[g][f];
      q0 += w * sVin[wv][f][0]; q1 += w * sVin[wv][f][1]; q2 += w * sVin[wv][f][2];
      k0 += u * sVin[wv][f][0]; k1 += u * sVin[wv][f][1]; k2 += u * sVin[wv][f][2];
    }
    float dot = q0 * k0 + q1 * k1 + q2 * k2;
    float ksq = k0 * k0 + k1 * k1 + k2 * k2 + EPS_;
    if (dot < 0.f) { float s = dot / ksq; q0 -= s * k0; q1 -= s * k1; q2 -= s * k2; }
    sVh[wv][g][0] = q0; sVh[wv][g][1] = q1; sVh[wv][g][2] = q2;
  }
  __syncthreads();
  if (ok && lane < 16) {
    float o0 = 0, o1 = 0, o2 = 0;
#pragma unroll
    for (int f = 0; f < 32; ++f) {
      float w = sW2[lane][f];
      o0 += w * sVh[wv][f][0]; o1 += w * sVh[wv][f][1]; o2 += w * sVh[wv][f][2];
    }
    size_t b = (size_t)nId * 48 + lane * 3;
    out_v[b + 0] = vA[b + 0] + o0;
    out_v[b + 1] = vA[b + 1] + o1;
    out_v[b + 2] = vA[b + 2] + o2;
  }
}

extern "C" void kernel_launch(void* const* d_in, const int* in_sizes, int n_in,
                              void* d_out, int out_size, void* d_ws, size_t ws_size,
                              hipStream_t stream) {
  const float* h_A    = (const float*)d_in[0];
  const float* v_A    = (const float*)d_in[1];
  const float* x_A    = (const float*)d_in[2];
  const float* h_B    = (const float*)d_in[3];
  const float* mask   = (const float*)d_in[4];
  const int*   src    = (const int*)d_in[5];
  const int*   dst    = (const int*)d_in[6];
  const float* rbf_k  = (const float*)d_in[7];
  const float* W_rbf1 = (const float*)d_in[8];
  const float* b_rbf1 = (const float*)d_in[9];
  const float* W_rbf2 = (const float*)d_in[10];
  const float* b_rbf2 = (const float*)d_in[11];
  const float* W_rbf3 = (const float*)d_in[12];
  const float* b_rbf3 = (const float*)d_in[13];
  const float* vn1_W  = (const float*)d_in[14];
  const float* vn1_U  = (const float*)d_in[15];
  const float* vn2_W  = (const float*)d_in[16];
  const float* vn2_U  = (const float*)d_in[17];
  const float* vn3_W  = (const float*)d_in[18];
  const float* vn3_U  = (const float*)d_in[19];
  const float* phi1_W = (const float*)d_in[20];
  const float* phi1_b = (const float*)d_in[21];
  const float* phi2_W = (const float*)d_in[22];
  const float* phi2_b = (const float*)d_in[23];
  const float* phi3_W = (const float*)d_in[24];
  const float* phi3_b = (const float*)d_in[25];
  const float* Wq     = (const float*)d_in[26];
  const float* Wk     = (const float*)d_in[27];
  const float* Wv     = (const float*)d_in[28];
  const float* bv     = (const float*)d_in[29];
  const float* gWih   = (const float*)d_in[30];
  const float* gWhh   = (const float*)d_in[31];
  const float* gbih   = (const float*)d_in[32];
  const float* gbhh   = (const float*)d_in[33];
  const float* vm_W1  = (const float*)d_in[34];
  const float* vm_U1  = (const float*)d_in[35];
  const float* vm_W2  = (const float*)d_in[36];

  // ---- workspace layout: f32 region then f16 region (~132 MB total) ----
  float* wsf = (float*)d_ws;
  size_t off = 0;
  float* vnorm1 = wsf + off; off += (size_t)NN * 16;
  float* vnorm2 = wsf + off; off += (size_t)NN * 16;
  float* vr3    = wsf + off; off += (size_t)NN * 48;
  float* hp     = wsf + off; off += (size_t)NN * 128;
  float* hpp    = wsf + off; off += (size_t)NN * 16;
  float* s3     = wsf + off; off += (size_t)NN * 16;
  float* vp     = wsf + off; off += (size_t)NN * 48;
  float* attnb  = wsf + off; off += (size_t)NN * 128;
  float* agg_h  = wsf + off; off += (size_t)NN * 128;
  float* agg_v  = wsf + off; off += (size_t)NN * 48;
  float* cnt    = wsf + off; off += (size_t)NN;
  off = (off + 15) & ~(size_t)15;  // 64B-align the f16 region

  _Float16* wsh = (_Float16*)(wsf + off);
  size_t hoff = 0;
  _Float16* q16     = wsh + hoff; hoff += (size_t)NN * 128;
  _Float16* kk16    = wsh + hoff; hoff += (size_t)NB_ * 128;
  _Float16* vvT16   = wsh + hoff; hoff += (size_t)DD * NB_;
  _Float16* phi1T16 = wsh + hoff; hoff += 128 * 160;
  _Float16* phi2T16 = wsh + hoff; hoff += 16 * 160;
  _Float16* phi3T16 = wsh + hoff; hoff += 16 * 128;
  _Float16* WqT16   = wsh + hoff; hoff += 128 * 128;
  _Float16* WkT16   = wsh + hoff; hoff += 128 * 128;
  _Float16* WvT16   = wsh + hoff; hoff += 128 * 128;
  _Float16* Wih16   = wsh + hoff; hoff += 384 * 256;
  _Float16* Whh16   = wsh + hoff; hoff += 384 * 128;

  float* out_v = (float*)d_out;               // [N,16,3]
  float* out_h = out_v + (size_t)NN * 48;     // [N,128]

  // 0) pack weights -> transposed, padded f16 (B-operand friendly)
  pack_wT_kernel<<<(128 * 160 + 255) / 256, 256, 0, stream>>>(phi1_W, phi1T16, 144, 128, 160);
  pack_wT_kernel<<<(16 * 160 + 255) / 256, 256, 0, stream>>>(phi2_W, phi2T16, 144, 16, 160);
  pack_wT_kernel<<<(16 * 128 + 255) / 256, 256, 0, stream>>>(phi3_W, phi3T16, 128, 16, 128);
  pack_wT_kernel<<<(128 * 128 + 255) / 256, 256, 0, stream>>>(Wq, WqT16, 128, 128, 128);
  pack_wT_kernel<<<(128 * 128 + 255) / 256, 256, 0, stream>>>(Wk, WkT16, 128, 128, 128);
  pack_wT_kernel<<<(128 * 128 + 255) / 256, 256, 0, stream>>>(Wv, WvT16, 128, 128, 128);
  pack_w_kernel<<<(384 * 256 + 255) / 256, 256, 0, stream>>>(gWih, Wih16, 256, 384, 256);
  pack_w_kernel<<<(384 * 128 + 255) / 256, 256, 0, stream>>>(gWhh, Whh16, 128, 384, 128);

  // 1) zero aggregation buffers (atomics accumulate; must reset each call)
  zero_kernel<<<(NN * DD + 255) / 256, 256, 0, stream>>>(agg_h, agg_v, cnt);

  // 2) per-node vector-neuron features
  node_vn_kernel<<<(NN + 255) / 256, 256, 0, stream>>>(
      v_A, vn1_W, vn1_U, vn2_W, vn2_U, vn3_W, vn3_U, vnorm1, vnorm2, vr3);

  // 3) WMMA GEMMs (templates: KTOT, K1, LDA1, LDA2, DOUT, LDO, bias, act, omode)
  // hp = lrelu([h_A, vnorm1] @ phi1_W + b)   [N,144]x[144,128]
  gemm_t<144, 128, 128, 16, 128, 128, true, true, 0>
      <<<dim3(NN / 16, 8), 32, 0, stream>>>(h_A, vnorm1, phi1T16, phi1_b, hp);
  // hpp = lrelu([h_A, vnorm2] @ phi2_W + b)  [N,144]x[144,16]
  gemm_t<144, 128, 128, 16, 16, 16, true, true, 0>
      <<<dim3(NN / 16, 1), 32, 0, stream>>>(h_A, vnorm2, phi2T16, phi2_b, hpp);
  // s3 = lrelu(h_A @ phi3_W + b)             [N,128]x[128,16]
  gemm_t<128, 128, 128, 128, 16, 16, true, true, 0>
      <<<dim3(NN / 16, 1), 32, 0, stream>>>(h_A, h_A, phi3T16, phi3_b, s3);
  // q16 = lrelu(h_A @ Wq)  (f16 row-major for attention A operand)
  gemm_t<128, 128, 128, 128, 128, 128, false, true, 1>
      <<<dim3(NN / 16, 8), 32, 0, stream>>>(h_A, h_A, WqT16, nullptr, q16);
  // kk16 = lrelu(h_B @ Wk)  (f16 row-major == K^T for scores B operand)
  gemm_t<128, 128, 128, 128, 128, 128, false, true, 1>
      <<<dim3(NB_ / 16, 8), 32, 0, stream>>>(h_B, h_B, WkT16, nullptr, kk16);
  // vvT16 = (h_B @ Wv + bv)^T  (f16 transposed for PV B operand)
  gemm_t<128, 128, 128, 128, 128, 256, true, false, 2>
      <<<dim3(NB_ / 16, 8), 32, 0, stream>>>(h_B, h_B, WvT16, bv, vvT16);

  // 4) vp = vn_relu3(v_A) * s3
  vp_mul_kernel<<<(NN * 48 + 255) / 256, 256, 0, stream>>>(vr3, s3, vp);

  // 5) cross attention (fused QK^T -> softmax -> PV)
  attn_kernel<<<NN / 16, 256, 0, stream>>>(q16, kk16, vvT16, mask, attnb);

  // 6) fused edge pass (L2-resident gathers + atomic scatter)
  edge_kernel<<<(EE + 7) / 8, 256, 0, stream>>>(
      x_A, src, dst, rbf_k, W_rbf1, b_rbf1, W_rbf2, b_rbf2, W_rbf3, b_rbf3,
      hp, hpp, vp, agg_h, agg_v, cnt);

  // 7) mean normalization of aggregates
  norm_agg_kernel<<<(NN * (DD + 48) + 255) / 256, 256, 0, stream>>>(agg_h, agg_v, cnt);

  // 8) fused GRU (WMMA gi/gh in LDS) -> node_upd
  gru_kernel<<<NN / 16, 256, 0, stream>>>(
      h_A, attnb, agg_h, Wih16, Whh16, gbih, gbhh, out_h);

  // 9) vector MLP residual -> v_evolved
  vmlp_kernel<<<(NN + 7) / 8, 256, 0, stream>>>(
      v_A, agg_v, vm_W1, vm_U1, vm_W2, out_v);

  (void)in_sizes; (void)n_in; (void)out_size; (void)ws_size;
}
